// RecursiveAttnPooling_37117107372369
// MI455X (gfx1250) — matrix operand order repaired and checked by reading, not compiled, not run on hardware
//
#include <hip/hip_runtime.h>
#include <hip/hip_bf16.h>

// ---------------------------------------------------------------------------
// RecursiveAttnPooling on MI455X (gfx1250, wave32, WMMA)
//
// Strategy: the only heavy tensors are h [16,8192,256] (134 MB) and the
// would-be intermediates base/out (134 MB each). We never materialize them:
// per iteration one fused kernel streams h once (async global->LDS), does
// BOTH 256x256 GEMMs with v_wmma_f32_16x16x32_f16 (f32 accum) and all
// T-reductions in-register/LDS. Only the softmax-weighted C update needs a
// second (light) pass over h. Total traffic ~1.2 GB -> ~52us at 23.3 TB/s.
// GEMMs run in two 8-ntile output halves (VGPR pressure ~170 < 256) with an
// explicit depth-2 software pipeline on the B-fragment loads so the WGP$ hit
// latency is covered by two in-flight WMMAs.
// ---------------------------------------------------------------------------

#define B_  16
#define T_  8192
#define D_  256
#define DP_ 256
#define E_  192
#define NIT 4

typedef __attribute__((ext_vector_type(16))) _Float16 v16h;
typedef __attribute__((ext_vector_type(8)))  float    v8f;

// gfx1250 async global->LDS copy (ASYNCcnt-tracked, no VGPR round trip).
#if defined(__has_builtin)
#if __has_builtin(__builtin_amdgcn_global_load_async_to_lds_b128)
#define HAVE_ASYNC_LDS 1
#endif
#endif

#if defined(HAVE_ASYNC_LDS)
typedef int v4i __attribute__((vector_size(16)));
typedef __attribute__((address_space(1))) v4i gas_v4i;   // global
typedef __attribute__((address_space(3))) v4i las_v4i;   // LDS
#endif

// ---------------------------------------------------------------------------
// zero helper
__global__ __launch_bounds__(256) void zero_kernel(float* __restrict__ p, int n) {
    int i = blockIdx.x * 256 + threadIdx.x;
    if (i < n) p[i] = 0.0f;
}

// ---------------------------------------------------------------------------
// Pass 0a: per-(b,d) sums of h and h^2 over T (for mu0/sig0)
__global__ __launch_bounds__(256) void stats_kernel(const float* __restrict__ h,
                                                    float* __restrict__ musum,
                                                    float* __restrict__ m2sum) {
    const int b  = blockIdx.y;
    const int t0 = blockIdx.x * 512;
    const int d  = threadIdx.x;
    float s1 = 0.0f, s2 = 0.0f;
    const float* p = h + ((size_t)b * T_ + t0) * D_ + d;
    for (int t = 0; t < 512; ++t) {
        float v = p[(size_t)t * D_];
        s1 += v;
        s2 += v * v;
    }
    atomicAdd(&musum[b * D_ + d], s1);
    atomicAdd(&m2sum[b * D_ + d], s2);
}

// ---------------------------------------------------------------------------
// Pass 0b: bias0[b,p] = mu0 @ W1[D:2D] + sig0 @ W1[2D:3D]; bias1 = bias0 (C0=0)
__global__ __launch_bounds__(256) void finalize_stats_kernel(
        const float* __restrict__ musum, const float* __restrict__ m2sum,
        const float* __restrict__ W1,
        float* __restrict__ bias0, float* __restrict__ bias1) {
    __shared__ float mu0s[D_];
    __shared__ float sg0s[D_];
    const int tid = threadIdx.x;
    for (int b = 0; b < B_; ++b) {
        {
            float mu  = musum[b * D_ + tid] * (1.0f / (float)T_);
            float m2  = m2sum[b * D_ + tid] * (1.0f / (float)T_);
            float var = m2 - mu * mu;
            if (var < 1e-8f) var = 1e-8f;
            mu0s[tid] = mu;
            sg0s[tid] = sqrtf(var);
        }
        __syncthreads();
        float acc = 0.0f;
        for (int dd = 0; dd < D_; ++dd) {
            acc += mu0s[dd] * W1[(size_t)(D_     + dd) * DP_ + tid];
            acc += sg0s[dd] * W1[(size_t)(2 * D_ + dd) * DP_ + tid];
        }
        bias0[b * DP_ + tid] = acc;
        bias1[b * DP_ + tid] = acc;   // iteration 0: C == 0 -> C@Wc == 0
        __syncthreads();
    }
}

// ---------------------------------------------------------------------------
// Pack W1a (=W1 rows 0..255) and W2 into f16, fragment-major B-operand layout:
//   frag = kk*16+nt holds the 32x16 block (K = kk*32.., N = nt*16..)
//   element (lane, i): K = kk*32 + 16*(i/8) + 8*(lane/16) + i%8 ; N = nt*16 + lane%16
// so each lane's 16 halves are contiguous -> two coalesced b128 loads.
__global__ __launch_bounds__(256) void pack_kernel(const float* __restrict__ W1,
                                                   const float* __restrict__ W2,
                                                   _Float16* __restrict__ W1pk,
                                                   _Float16* __restrict__ W2pk) {
    int gid  = blockIdx.x * 256 + threadIdx.x;   // 0 .. 131071
    int mat  = gid >> 16;
    int idx  = gid & 65535;
    int frag = idx >> 9;
    int r    = idx & 511;
    int lane = r >> 4;
    int i    = r & 15;
    int kk   = frag >> 4;
    int nt   = frag & 15;
    int K    = kk * 32 + ((i >> 3) << 4) + ((lane >> 4) << 3) + (i & 7);
    int N    = nt * 16 + (lane & 15);
    if (mat == 0) W1pk[idx] = (_Float16)W1[(size_t)K * DP_ + N];
    else          W2pk[idx] = (_Float16)W2[(size_t)K * D_  + N];
}

// ---------------------------------------------------------------------------
// Fused per-iteration kernel. Grid: (T/64, B). Block: 128 (= 4 waves).
// Each wave owns a 16-row strip: GEMM1 (h@W1a)+bias+relu -> LDS f16 ->
// GEMM2 (@W2)+b2 -> reductions (row-mean, out*wp, out*h, out*h^2).
__global__ __launch_bounds__(128) void fused_iter_kernel(
        const float*   __restrict__ h,
        const _Float16* __restrict__ W1pk,
        const _Float16* __restrict__ W2pk,
        const float*   __restrict__ bias1,   // [B,DP]  bias0 + C@Wc
        const float*   __restrict__ b2,      // [D]
        const float*   __restrict__ wp,      // [D]
        float*         __restrict__ sA,      // [B,T] row means (later softmaxed)
        float*         __restrict__ mu_p,    // [B,D] atomic accum
        float*         __restrict__ m2_p,    // [B,D] atomic accum
        float*         __restrict__ wpdot)   // [1]   atomic accum
{
    // 96 KB static LDS (gfx1250 allows up to 320 KB/WG) -> 3 WGs per WGP
    __shared__ float    hs[64 * D_];          // h tile, f32 (64 KB)
    __shared__ _Float16 rsc[4 * 16 * DP_];    // per-wave relu scratch (32 KB)

    const int b    = blockIdx.y;
    const int t0   = blockIdx.x * 64;
    const int tid  = threadIdx.x;
    const int wave = tid >> 5;          // wave32
    const int lane = tid & 31;
    const int ln16 = lane & 15;
    const int hi   = lane >> 4;         // 0 | 1
    const int mtop = hi * 8;

    // cooperative h tile load (64 rows x 256 f32, contiguous 64 KB)
    {
        const float4* src = (const float4*)(h + ((size_t)b * T_ + t0) * D_);
        float4*       dst = (float4*)hs;
#if defined(HAVE_ASYNC_LDS)
        for (int i = tid; i < 64 * 64; i += 128) {
            __builtin_amdgcn_global_load_async_to_lds_b128(
                (gas_v4i*)(uintptr_t)(src + i),
                (las_v4i*)(unsigned)(uintptr_t)(dst + i), 0, 0);
        }
#if __has_builtin(__builtin_amdgcn_s_wait_asynccnt)
        __builtin_amdgcn_s_wait_asynccnt(0);
#else
        asm volatile("s_wait_asynccnt 0x0" ::: "memory");
#endif
#else
        for (int i = tid; i < 64 * 64; i += 128) dst[i] = src[i];
#endif
    }
    __syncthreads();

    const float* hstrip = hs + wave * 16 * D_;
    _Float16*    scr    = rsc + wave * 16 * DP_;

    // ---- GEMM1: X = h_strip @ W1a, two 8-ntile output halves ---------------
    // B fragments software-pipelined at depth 2 (load frag f+2 before wmma f).
#pragma unroll 1
    for (int half = 0; half < 2; ++half) {
        const _Float16* WL = W1pk + (size_t)half * 8 * 512 + lane * 16;
        // fragment (kk, j): WL + (kk*16 + j)*512
        v8f acc[8];
#pragma unroll
        for (int j = 0; j < 8; ++j) acc[j] = v8f{};

        v16h a;
        {
            const float* row = hstrip + ln16 * D_;
#pragma unroll
            for (int i = 0; i < 16; ++i) {
                int k = ((i >> 3) << 4) + (hi << 3) + (i & 7);   // A-layout K map
                a[i] = (_Float16)row[k];
            }
        }
        v16h bA = *(const v16h*)(WL);
        v16h bB = *(const v16h*)(WL + 512);

#pragma unroll
        for (int kk = 0; kk < 8; ++kk) {
#pragma unroll
            for (int j = 0; j < 8; ++j) {
                v16h bc = bA;
                bA = bB;
                int nj  = j + 2;
                int nkk = kk + (nj >> 3);
                if (nkk < 8)
                    bB = *(const v16h*)(WL + ((size_t)nkk * 16 + (nj & 7)) * 512);
                acc[j] = __builtin_amdgcn_wmma_f32_16x16x32_f16(
                    false, a, false, bc, (short)0, acc[j], false, false);
            }
            if (kk < 7) {
                const float* row = hstrip + ln16 * D_ + (kk + 1) * 32;
#pragma unroll
                for (int i = 0; i < 16; ++i) {
                    int k = ((i >> 3) << 4) + (hi << 3) + (i & 7);
                    a[i] = (_Float16)row[k];
                }
            }
        }

        // bias + relu -> per-wave LDS scratch (C-layout -> row-major f16)
#pragma unroll
        for (int j = 0; j < 8; ++j) {
            int   d  = (half * 8 + j) * 16 + ln16;
            float bv = bias1[b * DP_ + d];
#pragma unroll
            for (int g = 0; g < 8; ++g) {
                float v = acc[j][g] + bv;
                v = v > 0.0f ? v : 0.0f;
                scr[(mtop + g) * DP_ + d] = (_Float16)v;
            }
        }
    }
    // same-wave LDS RAW: compiler inserts s_wait_dscnt, no barrier needed

    // ---- GEMM2 + epilogue, two 8-ntile output halves ------------------------
    float rs[8];
#pragma unroll
    for (int g = 0; g < 8; ++g) rs[g] = 0.0f;
    float wpacc = 0.0f;

#pragma unroll 1
    for (int half = 0; half < 2; ++half) {
        const _Float16* WL = W2pk + (size_t)half * 8 * 512 + lane * 16;
        v8f acc[8];
#pragma unroll
        for (int j = 0; j < 8; ++j) acc[j] = v8f{};

        v16h a;
        {
            const _Float16* row = scr + ln16 * DP_;
#pragma unroll
            for (int i = 0; i < 16; ++i) {
                int k = ((i >> 3) << 4) + (hi << 3) + (i & 7);
                a[i] = row[k];
            }
        }
        v16h bA = *(const v16h*)(WL);
        v16h bB = *(const v16h*)(WL + 512);

#pragma unroll
        for (int kk = 0; kk < 8; ++kk) {
#pragma unroll
            for (int j = 0; j < 8; ++j) {
                v16h bc = bA;
                bA = bB;
                int nj  = j + 2;
                int nkk = kk + (nj >> 3);
                if (nkk < 8)
                    bB = *(const v16h*)(WL + ((size_t)nkk * 16 + (nj & 7)) * 512);
                acc[j] = __builtin_amdgcn_wmma_f32_16x16x32_f16(
                    false, a, false, bc, (short)0, acc[j], false, false);
            }
            if (kk < 7) {
                const _Float16* row = scr + ln16 * DP_ + (kk + 1) * 32;
#pragma unroll
                for (int i = 0; i < 16; ++i) {
                    int k = ((i >> 3) << 4) + (hi << 3) + (i & 7);
                    a[i] = row[k];
                }
            }
        }

        // epilogue for these 8 ntiles: out = acc + b2; reduce against h tile
#pragma unroll
        for (int j = 0; j < 8; ++j) {
            int   d    = (half * 8 + j) * 16 + ln16;
            float b2v  = b2[d];
            float wpv  = wp[d];
            float mu_l = 0.0f, m2_l = 0.0f;
#pragma unroll
            for (int g = 0; g < 8; ++g) {
                float o  = acc[j][g] + b2v;            // out[t, d]
                float hv = hstrip[(mtop + g) * D_ + d];
                rs[g]  += o;
                wpacc  += o * wpv;
                mu_l   += o * hv;
                m2_l   += o * hv * hv;
            }
            // lanes l and l+16 hold the same d -> cross-half-wave add
            mu_l += __shfl_xor(mu_l, 16);
            m2_l += __shfl_xor(m2_l, 16);
            if (hi == 0) {
                atomicAdd(&mu_p[b * D_ + d], mu_l);
                atomicAdd(&m2_p[b * D_ + d], m2_l);
            }
        }
    }

    // row means -> sA  (reduce across the 16 lanes of each half-wave)
#pragma unroll
    for (int g = 0; g < 8; ++g) {
        float v = rs[g];
        v += __shfl_xor(v, 1);
        v += __shfl_xor(v, 2);
        v += __shfl_xor(v, 4);
        v += __shfl_xor(v, 8);
        if (ln16 == 0) {
            int t = t0 + wave * 16 + mtop + g;
            sA[(size_t)b * T_ + t] = v * (1.0f / (float)D_);
        }
    }

    // sum(out * wp) over everything
    wpacc += __shfl_xor(wpacc, 1);
    wpacc += __shfl_xor(wpacc, 2);
    wpacc += __shfl_xor(wpacc, 4);
    wpacc += __shfl_xor(wpacc, 8);
    wpacc += __shfl_xor(wpacc, 16);
    if (lane == 0) atomicAdd(wpdot, wpacc);
}

// ---------------------------------------------------------------------------
// Softmax over T per batch, in place (sA: row-means -> normalized A)
__global__ __launch_bounds__(256) void softmax_kernel(float* __restrict__ s) {
    const int b   = blockIdx.x;
    const int tid = threadIdx.x;
    __shared__ float red[256];
    float* sb = s + (size_t)b * T_;

    float mx = -3.4e38f;
    for (int t = tid; t < T_; t += 256) mx = fmaxf(mx, sb[t]);
    red[tid] = mx;
    __syncthreads();
    for (int w = 128; w > 0; w >>= 1) {
        if (tid < w) red[tid] = fmaxf(red[tid], red[tid + w]);
        __syncthreads();
    }
    mx = red[0];
    __syncthreads();

    float sum = 0.0f;
    for (int t = tid; t < T_; t += 256) sum += expf(sb[t] - mx);
    red[tid] = sum;
    __syncthreads();
    for (int w = 128; w > 0; w >>= 1) {
        if (tid < w) red[tid] += red[tid + w];
        __syncthreads();
    }
    float inv = 1.0f / red[0];
    for (int t = tid; t < T_; t += 256) sb[t] = expf(sb[t] - mx) * inv;
}

// ---------------------------------------------------------------------------
// Csum[b,d] = sum_t A[b,t] * h[b,t,d]   (second, light pass over h)
__global__ __launch_bounds__(256) void cupd_kernel(const float* __restrict__ h,
                                                   const float* __restrict__ A,
                                                   float* __restrict__ Csum) {
    const int b  = blockIdx.y;
    const int t0 = blockIdx.x * 512;
    const int d  = threadIdx.x;
    float acc = 0.0f;
    const float* hp = h + ((size_t)b * T_ + t0) * D_ + d;
    const float* ap = A + (size_t)b * T_ + t0;
    for (int t = 0; t < 512; ++t) acc += ap[t] * hp[(size_t)t * D_];
    atomicAdd(&Csum[b * D_ + d], acc);
}

// ---------------------------------------------------------------------------
// Per-iteration tail: emb, p, C update, bias1 for next iteration
__global__ __launch_bounds__(256) void finalize_iter_kernel(
        const float* __restrict__ mu_p, const float* __restrict__ m2_p,
        const float* __restrict__ w0,   const float* __restrict__ b0,
        const float* __restrict__ wpd,  const float* __restrict__ bp,
        const float* __restrict__ Csum, const float* __restrict__ Wc,
        float* __restrict__ C,          const float* __restrict__ bias0,
        float* __restrict__ bias1,      float* __restrict__ out, int iter) {
    __shared__ float catv[2 * D_];
    __shared__ float Cn[D_];
    const int tid = threadIdx.x;
    for (int b = 0; b < B_; ++b) {
        {
            float mu  = mu_p[b * D_ + tid];
            float m2  = m2_p[b * D_ + tid];
            float var = m2 - mu * mu;
            if (var < 1e-8f) var = 1e-8f;
            catv[tid]      = mu;
            catv[D_ + tid] = sqrtf(var);
            float cn = C[b * D_ + tid] + Csum[b * D_ + tid] * (1.0f / (float)T_);
            Cn[tid] = cn;
            C[b * D_ + tid] = cn;
        }
        __syncthreads();
        if (tid < E_) {
            float acc = b0[tid];
            for (int k = 0; k < 2 * D_; ++k) acc += catv[k] * w0[(size_t)k * E_ + tid];
            out[b * (NIT * E_) + iter * E_ + tid] = acc;   // embeddings [B,N,E]
        }
        {
            float acc = bias0[b * DP_ + tid];
            for (int dd = 0; dd < D_; ++dd) acc += Cn[dd] * Wc[(size_t)dd * DP_ + tid];
            bias1[b * DP_ + tid] = acc;                    // for next iteration
        }
        __syncthreads();
    }
    if (tid == 0) {
        float arg = -wpd[0] * (1.0f / ((float)B_ * (float)T_)) + bp[0];
        out[B_ * NIT * E_ + iter] = 1.0f / (1.0f + expf(-arg));
    }
}

// ---------------------------------------------------------------------------
extern "C" void kernel_launch(void* const* d_in, const int* in_sizes, int n_in,
                              void* d_out, int out_size, void* d_ws, size_t ws_size,
                              hipStream_t stream) {
    const float* h  = (const float*)d_in[0];
    const float* W1 = (const float*)d_in[1];
    const float* Wc = (const float*)d_in[2];
    const float* W2 = (const float*)d_in[3];
    const float* b2 = (const float*)d_in[4];
    const float* wp = (const float*)d_in[5];
    const float* bp = (const float*)d_in[6];
    const float* w0 = (const float*)d_in[7];
    const float* b0 = (const float*)d_in[8];
    float* out = (float*)d_out;

    // workspace carve (floats); all offsets keep >=32B alignment
    float* wsf   = (float*)d_ws;
    float* sA    = wsf;                 // [B*T]   = 131072
    float* musum = wsf + 131072;        // [B*D]
    float* m2sum = musum + 4096;
    float* C     = m2sum + 4096;
    float* Csum  = C + 4096;
    float* mu_p  = Csum + 4096;
    float* m2_p  = mu_p + 4096;
    float* wpd   = m2_p + 4096;         // 64 (padded)
    float* bias0 = wpd + 64;
    float* bias1 = bias0 + 4096;
    _Float16* W1pk = (_Float16*)(bias1 + 4096);   // 65536 halves
    _Float16* W2pk = W1pk + 65536;                // 65536 halves

    // init: musum, m2sum, C are contiguous -> one zero sweep
    zero_kernel<<<dim3((3 * 4096 + 255) / 256), dim3(256), 0, stream>>>(musum, 3 * 4096);
    stats_kernel<<<dim3(T_ / 512, B_), dim3(256), 0, stream>>>(h, musum, m2sum);
    finalize_stats_kernel<<<dim3(1), dim3(256), 0, stream>>>(musum, m2sum, W1, bias0, bias1);
    pack_kernel<<<dim3(512), dim3(256), 0, stream>>>(W1, W2, W1pk, W2pk);

    for (int it = 0; it < NIT; ++it) {
        // Csum, mu_p, m2_p, wpd are contiguous -> one zero sweep
        zero_kernel<<<dim3((3 * 4096 + 64 + 255) / 256), dim3(256), 0, stream>>>(Csum, 3 * 4096 + 64);
        fused_iter_kernel<<<dim3(T_ / 64, B_), dim3(128), 0, stream>>>(
            h, W1pk, W2pk, bias1, b2, wp, sA, mu_p, m2_p, wpd);
        softmax_kernel<<<dim3(B_), dim3(256), 0, stream>>>(sA);
        cupd_kernel<<<dim3(T_ / 512, B_), dim3(256), 0, stream>>>(h, sA, Csum);
        finalize_iter_kernel<<<dim3(1), dim3(256), 0, stream>>>(
            mu_p, m2_p, w0, b0, wpd, bp, Csum, Wc, C, bias0, bias1, out, it);
    }
}
